// RTier_35390530519884
// MI455X (gfx1250) — matrix operand order repaired and checked by reading, not compiled, not run on hardware
//
#include <hip/hip_runtime.h>
#include <stdint.h>

// ---------------------------------------------------------------------------
// GRU scan for MI455X (gfx1250), wave32 + WMMA bf16 + Tensor Data Mover.
//   4 workgroups x 512 threads (16 waves). WG b handles batch rows [16b,16b+16).
//   Wave w owns hidden columns [16w, 16w+16); computes r/z/n gate tiles and the
//   output-projection tile for those columns entirely in registers via
//   v_wmma_f32_16x16x32_bf16 (f32 accumulate), h carried in f32 in LDS.
//   x_{t+1} is prefetched into LDS by the TDM (tensor_load_to_lds, TENSORcnt)
//   while step t computes, hiding global latency on the sequential scan.
// ---------------------------------------------------------------------------

#define TSEQ    4096
#define NBATCH  64
#define DK      256      // B_DIM == H_DIM == R_DIM
#define G3      768      // 3*H (gates r|z|n)
#define XPAD    264      // bf16 LDS row stride (elements) - bank-conflict pad
#define FPAD    264      // f32 LDS row stride (elements)

typedef __bf16 bf16_t;
typedef bf16_t bf16x16 __attribute__((ext_vector_type(16)));
typedef bf16_t bf16x2  __attribute__((ext_vector_type(2)));
typedef float  f32x8   __attribute__((ext_vector_type(8)));
typedef unsigned int u32x4 __attribute__((ext_vector_type(4)));
typedef int          i32x4 __attribute__((ext_vector_type(4)));
typedef int          i32x8 __attribute__((ext_vector_type(8)));

union Frag { bf16x16 v; uint4 q[2]; };

// Native bf16 converts (v_cvt_pk_bf16_f32 / v_fma_mixlo_bf16 on gfx1250), RNE.
static __device__ __forceinline__ unsigned short f2bf(float f) {
    return __builtin_bit_cast(unsigned short, (__bf16)f);
}
static __device__ __forceinline__ unsigned int pack_bf2(float lo, float hi) {
    bf16x2 v; v[0] = (__bf16)lo; v[1] = (__bf16)hi;
    return __builtin_bit_cast(unsigned int, v);
}
// TRANS-unit-only activations (v_exp_f32 + v_rcp_f32, no v_div_fixup chains);
// these co-issue with the XDL/WMMA pipe. ~1 ulp rcp error << bf16 GEMM noise.
static __device__ __forceinline__ float fast_sigmoid(float x) {
    return __builtin_amdgcn_rcpf(1.0f + __expf(-x));
}
static __device__ __forceinline__ float fast_tanh(float x) {
    return 2.0f * __builtin_amdgcn_rcpf(1.0f + __expf(-2.0f * x)) - 1.0f;
}

// ---------------------------------------------------------------------------
// TDM: async DMA of a 16-row x 256-col f32 tile (row stride TSEQ*DK elements)
// from global memory into LDS. D# built per CDNA5 ISA 8.3/8.4 (group0/group1);
// groups 2/3 zero => 2D tensor. Tracked by TENSORcnt.
// ---------------------------------------------------------------------------
static __device__ __forceinline__ void tdm_load_x_tile(const float* gsrc,
                                                       unsigned lds_byte_addr) {
    unsigned long long ga = (unsigned long long)(uintptr_t)gsrc;
    u32x4 g0;
    g0[0] = 1u;                                           // count=1 (valid), user
    g0[1] = lds_byte_addr;                                // lds_addr (bytes)
    g0[2] = (unsigned)(ga & 0xFFFFFFFFu);                 // global_addr[31:0]
    g0[3] = (unsigned)((ga >> 32) & 0x01FFFFFFu)          // global_addr[56:32]
          | (2u << 30);                                   // type=2 ("image")
    i32x8 g1;
    g1[0] = (int)(2u << 16);          // workgroup_mask=0, data_size=2 (4 bytes)
    g1[1] = (int)((unsigned)DK << 16);        // tensor_dim0 = 256 (lo16 @ b63:48)
    g1[2] = (int)(16u << 16);                 // tensor_dim0 hi=0 | tensor_dim1=16
    g1[3] = (int)((unsigned)DK << 16);        // tensor_dim1 hi=0 | tile_dim0=256
    g1[4] = 16;                               // tile_dim1=16, tile_dim2=0
    g1[5] = TSEQ * DK;                        // tensor_dim0_stride lo32 (elements)
    g1[6] = 0;                                // stride hi16 | tensor_dim1_stride lo
    g1[7] = 0;
    i32x4 gz = {0, 0, 0, 0};
#if __clang_major__ >= 23
    i32x8 z8 = {0, 0, 0, 0, 0, 0, 0, 0};
    __builtin_amdgcn_tensor_load_to_lds(g0, g1, gz, gz, z8, 0);
#else
    __builtin_amdgcn_tensor_load_to_lds(g0, g1, gz, gz, 0);
#endif
}

// ---------------------------------------------------------------------------
// Weight prep: transpose f32 [K][N] -> bf16 [N][K] so B-operand fragments are
// contiguous 16B chunks per lane (matches CDNA5 32x16 bf16 B-matrix layout).
// ---------------------------------------------------------------------------
__global__ void gru_prep_weights(const float* __restrict__ Wx,
                                 const float* __restrict__ Wh,
                                 const float* __restrict__ Wo,
                                 unsigned short* __restrict__ WxT,
                                 unsigned short* __restrict__ WhT,
                                 unsigned short* __restrict__ WoT) {
    int i = blockIdx.x * blockDim.x + threadIdx.x;
    if (i < G3 * DK) {                    // [768][256] <- [256][768]
        int n = i / DK, k = i % DK;
        WxT[i] = f2bf(Wx[(size_t)k * G3 + n]);
        WhT[i] = f2bf(Wh[(size_t)k * G3 + n]);
    }
    if (i < DK * DK) {                    // [256][256] <- [256][256]
        int n = i / DK, k = i % DK;
        WoT[i] = f2bf(Wo[(size_t)k * DK + n]);
    }
}

// ---------------------------------------------------------------------------
// The sequential scan kernel.
// ---------------------------------------------------------------------------
__global__ void __launch_bounds__(512, 1)
gru_scan(const float* __restrict__ b_batch,      // [64][4096][256] f32
         const float* __restrict__ is_summ,      // [64][4096] f32
         const unsigned short* __restrict__ WxT, // [768][256] bf16
         const unsigned short* __restrict__ WhT, // [768][256] bf16
         const unsigned short* __restrict__ WoT, // [256][256] bf16
         const float* __restrict__ bx,           // [768]
         const float* __restrict__ bh,           // [768]
         const float* __restrict__ bo,           // [256]
         float* __restrict__ out,                // [64][4096][256]
         float* __restrict__ h_last)             // [64][256]
{
    __shared__ float sXf[2][16 * DK];                // TDM landing pads (f32)
    __shared__ unsigned short sXt[16 * XPAD];        // x_t tile, bf16
    __shared__ unsigned short sHbf[2][16 * XPAD];    // h state bf16, ping-pong
    __shared__ unsigned short sHo[16 * XPAD];        // h_new (pre-reset) bf16
    __shared__ float sH32[16 * FPAD];                // h state f32 (carry)
    __shared__ float sS[16];                         // is_summ per row
    __shared__ float sBx[G3], sBh[G3], sBo[DK];

    const int tid  = threadIdx.x;
    const int wave = tid >> 5;
    const int lane = tid & 31;
    const int nl   = lane & 15;     // local column within 16-wide tile
    const int half = lane >> 4;     // lane half (A: K-half, B: K-half, C: M-half)
    const int b0   = blockIdx.x * 16;
    const int hc0  = wave * 16;     // hidden-column tile base for this wave

    // ---- one-time staging ----
    for (int i = tid; i < G3; i += 512) { sBx[i] = bx[i]; sBh[i] = bh[i]; }
    for (int i = tid; i < DK; i += 512) sBo[i] = bo[i];
    for (int i = tid; i < 16 * XPAD; i += 512) { sHbf[0][i] = 0; sHbf[1][i] = 0; sHo[i] = 0; }
    for (int i = tid; i < 16 * FPAD; i += 512) sH32[i] = 0.0f;
    __syncthreads();

    const float* xbase = b_batch + (size_t)b0 * TSEQ * DK;  // tile row 0, t=0
    if (wave == 0) {
        // prologue: kick off DMA of x_0
        tdm_load_x_tile(xbase, (unsigned)(uintptr_t)&sXf[0][0]);
    }

    // per-lane weight row pointers ([N][K] bf16); half selects K 0-7 & 16-23 vs 8-15 & 24-31
    const unsigned short* pBxR = WxT + (size_t)(      hc0 + nl) * DK + half * 8;
    const unsigned short* pBxZ = WxT + (size_t)(256 + hc0 + nl) * DK + half * 8;
    const unsigned short* pBxN = WxT + (size_t)(512 + hc0 + nl) * DK + half * 8;
    const unsigned short* pBhR = WhT + (size_t)(      hc0 + nl) * DK + half * 8;
    const unsigned short* pBhZ = WhT + (size_t)(256 + hc0 + nl) * DK + half * 8;
    const unsigned short* pBhN = WhT + (size_t)(512 + hc0 + nl) * DK + half * 8;
    const unsigned short* pBO  = WoT + (size_t)(      hc0 + nl) * DK + half * 8;

    const float bR  = sBx[      hc0 + nl] + sBh[      hc0 + nl];
    const float bZ  = sBx[256 + hc0 + nl] + sBh[256 + hc0 + nl];
    const float bXn = sBx[512 + hc0 + nl];
    const float bHn = sBh[512 + hc0 + nl];
    const float bO  = sBo[hc0 + nl];

    // Output base pointers for row pairs (m = half*8 + 2rr, +1). Adjacent rows
    // are TSEQ*DK elements (4 MB) apart, which fits the signed 24-bit ioffset,
    // so each pair costs one pointer; advance by DK elements per step.
    // Named scalars (not an array) so addrspace inference keeps global stores.
    float* pO0 = out + (size_t)(b0 + half * 8 + 0) * TSEQ * DK + hc0 + nl;
    float* pO1 = out + (size_t)(b0 + half * 8 + 2) * TSEQ * DK + hc0 + nl;
    float* pO2 = out + (size_t)(b0 + half * 8 + 4) * TSEQ * DK + hc0 + nl;
    float* pO3 = out + (size_t)(b0 + half * 8 + 6) * TSEQ * DK + hc0 + nl;

    for (int t = 0; t < TSEQ; ++t) {
        const int cur = t & 1, prv = cur ^ 1;

        // ---- phase 0: wait for x_t DMA, launch x_{t+1} DMA, convert to bf16 --
        if (wave == 0) {
            __builtin_amdgcn_s_wait_tensorcnt(0);      // x_t tile landed in LDS
            if (t + 1 < TSEQ)
                tdm_load_x_tile(xbase + (size_t)(t + 1) * DK,
                                (unsigned)(uintptr_t)&sXf[cur ^ 1][0]);
        }
        __syncthreads();                                // publish TDM data to WG
        {
            const int m = tid >> 5;              // row 0..15
            const int k = (tid & 31) * 8;        // col base 0..248
            const float* src = &sXf[cur][m * DK + k];
            float4 f0 = *(const float4*)(src);
            float4 f1 = *(const float4*)(src + 4);
            uint4 pk;
            pk.x = pack_bf2(f0.x, f0.y);
            pk.y = pack_bf2(f0.z, f0.w);
            pk.z = pack_bf2(f1.x, f1.y);
            pk.w = pack_bf2(f1.z, f1.w);
            *(uint4*)(sXt + m * XPAD + k) = pk;
            if (tid < 16) sS[tid] = is_summ[(size_t)(b0 + tid) * TSEQ + t];
        }
        __syncthreads();

        // ---- phase 1: gate GEMMs (K=256, 8 x k32 steps) ----
        f32x8 accR = {}, accZ = {}, accNx = {}, accNh = {};
        const unsigned short* aXrow = sXt        + nl * XPAD + half * 8;
        const unsigned short* aHrow = sHbf[prv]  + nl * XPAD + half * 8;
        for (int kk = 0; kk < 8; ++kk) {
            const int k0 = kk * 32;
            Frag ax, ah, bq;
            // A fragments: 16x32 bf16 layout (K 0-7 & 16-23 / K 8-15 & 24-31)
            ax.q[0] = *(const uint4*)(aXrow + k0);
            ax.q[1] = *(const uint4*)(aXrow + k0 + 16);
            ah.q[0] = *(const uint4*)(aHrow + k0);
            ah.q[1] = *(const uint4*)(aHrow + k0 + 16);
            // r-gate: gx_r + gh_r accumulated into one C
            bq.q[0] = *(const uint4*)(pBxR + k0); bq.q[1] = *(const uint4*)(pBxR + k0 + 16);
            accR = __builtin_amdgcn_wmma_f32_16x16x32_bf16(false, ax.v, false, bq.v, (short)0, accR, false, false);
            bq.q[0] = *(const uint4*)(pBhR + k0); bq.q[1] = *(const uint4*)(pBhR + k0 + 16);
            accR = __builtin_amdgcn_wmma_f32_16x16x32_bf16(false, ah.v, false, bq.v, (short)0, accR, false, false);
            // z-gate: gx_z + gh_z
            bq.q[0] = *(const uint4*)(pBxZ + k0); bq.q[1] = *(const uint4*)(pBxZ + k0 + 16);
            accZ = __builtin_amdgcn_wmma_f32_16x16x32_bf16(false, ax.v, false, bq.v, (short)0, accZ, false, false);
            bq.q[0] = *(const uint4*)(pBhZ + k0); bq.q[1] = *(const uint4*)(pBhZ + k0 + 16);
            accZ = __builtin_amdgcn_wmma_f32_16x16x32_bf16(false, ah.v, false, bq.v, (short)0, accZ, false, false);
            // n-gate: gx_n and gh_n kept separate (n = tanh(gx_n + r*gh_n))
            bq.q[0] = *(const uint4*)(pBxN + k0); bq.q[1] = *(const uint4*)(pBxN + k0 + 16);
            accNx = __builtin_amdgcn_wmma_f32_16x16x32_bf16(false, ax.v, false, bq.v, (short)0, accNx, false, false);
            bq.q[0] = *(const uint4*)(pBhN + k0); bq.q[1] = *(const uint4*)(pBhN + k0 + 16);
            accNh = __builtin_amdgcn_wmma_f32_16x16x32_bf16(false, ah.v, false, bq.v, (short)0, accNh, false, false);
        }

        // ---- gate nonlinearity + h update, all in-register ----
        // C/D layout: lane half picks M 0-7 vs 8-15, vgpr r is row within half.
        #pragma unroll
        for (int r = 0; r < 8; ++r) {
            const int m = half * 8 + r;
            const float hprev = sH32[m * FPAD + hc0 + nl];
            const float rg = fast_sigmoid(accR[r] + bR);
            const float zg = fast_sigmoid(accZ[r] + bZ);
            const float ng = fast_tanh((accNx[r] + bXn) + rg * (accNh[r] + bHn));
            const float hn = (1.0f - zg) * ng + zg * hprev;     // h_new
            const float hx = (1.0f - sS[m]) * hn;               // boundary reset
            sHo[m * XPAD + hc0 + nl]       = f2bf(hn);          // for Wo GEMM
            sHbf[cur][m * XPAD + hc0 + nl] = f2bf(hx);          // next-step A
            sH32[m * FPAD + hc0 + nl]      = hx;                // f32 carry
        }
        __syncthreads();

        // ---- phase 2: output projection r_out = h_new @ Wo + bo ----
        f32x8 accO = {};
        const unsigned short* aOrow = sHo + nl * XPAD + half * 8;
        for (int kk = 0; kk < 8; ++kk) {
            const int k0 = kk * 32;
            Frag ao, bq;
            ao.q[0] = *(const uint4*)(aOrow + k0);
            ao.q[1] = *(const uint4*)(aOrow + k0 + 16);
            bq.q[0] = *(const uint4*)(pBO + k0);
            bq.q[1] = *(const uint4*)(pBO + k0 + 16);
            accO = __builtin_amdgcn_wmma_f32_16x16x32_bf16(false, ao.v, false, bq.v, (short)0, accO, false, false);
        }
        pO0[0]         = accO[0] + bO;      // ioffset 0
        pO0[TSEQ * DK] = accO[1] + bO;      // ioffset +4MB (imm24)
        pO1[0]         = accO[2] + bO;
        pO1[TSEQ * DK] = accO[3] + bO;
        pO2[0]         = accO[4] + bO;
        pO2[TSEQ * DK] = accO[5] + bO;
        pO3[0]         = accO[6] + bO;
        pO3[TSEQ * DK] = accO[7] + bO;
        pO0 += DK; pO1 += DK; pO2 += DK; pO3 += DK;   // next timestep
        // NOTE: no end-of-step barrier needed - the post-TDM and post-convert
        // barriers of step t+1 already separate phase-2 reads of sHo/sXt from
        // their next writers.
    }

    // ---- h_last ----
    __syncthreads();
    for (int i = tid; i < 16 * DK; i += 512) {
        const int m = i >> 8, k = i & 255;
        h_last[(size_t)(b0 + m) * DK + k] = sH32[m * FPAD + k];
    }
}

// ---------------------------------------------------------------------------
// Harness entry.
// ---------------------------------------------------------------------------
extern "C" void kernel_launch(void* const* d_in, const int* in_sizes, int n_in,
                              void* d_out, int out_size, void* d_ws, size_t ws_size,
                              hipStream_t stream) {
    (void)in_sizes; (void)n_in; (void)out_size; (void)ws_size;
    const float* b_batch = (const float*)d_in[0];
    const float* is_summ = (const float*)d_in[1];
    const float* Wx      = (const float*)d_in[2];
    const float* Wh      = (const float*)d_in[3];
    const float* bx      = (const float*)d_in[4];
    const float* bh      = (const float*)d_in[5];
    const float* Wo      = (const float*)d_in[6];
    const float* bo      = (const float*)d_in[7];

    unsigned short* WxT = (unsigned short*)d_ws;          // [768][256] bf16
    unsigned short* WhT = WxT + (size_t)G3 * DK;          // [768][256] bf16
    unsigned short* WoT = WhT + (size_t)G3 * DK;          // [256][256] bf16

    float* out    = (float*)d_out;                        // [64][4096][256]
    float* h_last = out + (size_t)NBATCH * TSEQ * DK;     // [64][256]

    gru_prep_weights<<<dim3((G3 * DK + 255) / 256), dim3(256), 0, stream>>>(
        Wx, Wh, Wo, WxT, WhT, WoT);
    gru_scan<<<dim3(NBATCH / 16), dim3(512), 0, stream>>>(
        b_batch, is_summ, WxT, WhT, WoT, bx, bh, bo, out, h_last);
}
